// RSTLayer_26560077758946
// MI455X (gfx1250) — compile-verified, matrix operand order checked by reading, not run on hardware
//
#include <hip/hip_runtime.h>
#include <hip/hip_bf16.h>
#include <math.h>
#include <stdint.h>

// ---------------- problem constants ----------------
#define Hd    2048
#define Sd    2048
#define Bd    2
#define SELd  512
#define MLPd  8192
#define Rd    64
#define NHd   16
#define DHd   128
#define EPSf  1e-5f
#define MROWS (Bd * Sd)                 // 4096
#define QSCALE 0.08838834764831845f    // 1/sqrt(128)

typedef __attribute__((ext_vector_type(16))) __bf16 v16bf;
typedef __attribute__((ext_vector_type(8)))  __bf16 v8bf;
typedef __attribute__((ext_vector_type(8)))  float  v8f;

__device__ __forceinline__ v16bf frag16(const __bf16* p) {
    v16bf r;
    ((v8bf*)&r)[0] = *(const v8bf*)(p);
    ((v8bf*)&r)[1] = *(const v8bf*)(p + 16);
    return r;
}

__device__ __forceinline__ unsigned short bfbits(__bf16 b) {
    union { __bf16 b; unsigned short s; } u; u.b = b; return u.s;
}

__device__ __forceinline__ float softplus_f(float x) {
    return (x > 20.f) ? x : log1pf(__expf(x));
}

// async copy: 16B per lane, global -> LDS, tracked by ASYNCcnt
__device__ __forceinline__ void async_b128(const void* gsrc, void* ldst) {
    unsigned long long ga = (unsigned long long)(uintptr_t)gsrc;
    unsigned           la = (unsigned)(uintptr_t)ldst;   // low 32 bits = LDS offset
    asm volatile("global_load_async_to_lds_b128 %0, %1, off" :: "v"(la), "v"(ga) : "memory");
}
__device__ __forceinline__ void wait_async0() {
    asm volatile("s_wait_asynccnt 0x0" ::: "memory");
}

// =====================================================================
// Weight transpose + convert: W[K][N] fp32 -> Wt[N][K] bf16
// =====================================================================
__global__ __launch_bounds__(256)
void transpose_conv_bf16(const float* __restrict__ W, __bf16* __restrict__ Wt,
                         int K, int N)
{
    __shared__ float tile[32][33];
    const int kb = blockIdx.x * 32;
    const int nb = blockIdx.y * 32;
    const int tx = threadIdx.x & 31;
    const int ty = threadIdx.x >> 5;
#pragma unroll
    for (int i = 0; i < 32; i += 8)
        tile[ty + i][tx] = W[(size_t)(kb + ty + i) * N + nb + tx];
    __syncthreads();
#pragma unroll
    for (int i = 0; i < 32; i += 8)
        Wt[(size_t)(nb + ty + i) * K + kb + tx] = (__bf16)tile[tx][ty + i];
}

__global__ __launch_bounds__(256)
void conv_bf16(const float* __restrict__ X, __bf16* __restrict__ Y)
{
    size_t i = (size_t)blockIdx.x * 256 + threadIdx.x;
    Y[i] = (__bf16)X[i];
}

// =====================================================================
// GEMM: C[M,N] = A[M,K](bf16) @ Wt[N,K](bf16 pre-transposed)
// Block 128 x (64*WNT), BK=32, 8 waves (4m x 2n), wave tile 32 x (32*WNT).
// Double-buffered LDS fed by GLOBAL_LOAD_ASYNC_TO_LDS_B128.
// mode: 0=store f32, 1=accum f32, 2=store bf16
// =====================================================================
#define LDAB 40   // padded LDS row stride (bf16): 80B = 5*16B keeps b128 alignment

template<int WNT>
__global__ __launch_bounds__(256)
void gemm_bf16_wmma(const __bf16* __restrict__ A, int lda,
                    const __bf16* __restrict__ Wt, int ldb,
                    void* __restrict__ Cv, int M, int N, int K, int mode)
{
    constexpr int BN = 64 * WNT;
    __shared__ __align__(16) __bf16 Als[2][128 * LDAB];
    __shared__ __align__(16) __bf16 Bls[2][BN * LDAB];

    const int t    = threadIdx.x;
    const int lane = t & 31;
    const int w    = t >> 5;
    const int wm   = w & 3;
    const int wn   = w >> 2;
    const int m0   = blockIdx.y * 128;
    const int n0   = blockIdx.x * BN;
    const int half = lane >> 4;
    const int l16  = lane & 15;
    const int kb8  = half * 8;

    // per-thread 16B staging chunk coordinates
    int arow[2], akc[2], brow[WNT], bkc[WNT];
#pragma unroll
    for (int i = 0; i < 2; ++i) {
        int f = t + i * 256; arow[i] = f >> 2; akc[i] = (f & 3) << 3;
    }
#pragma unroll
    for (int i = 0; i < WNT; ++i) {
        int f = t + i * 256; brow[i] = f >> 2; bkc[i] = (f & 3) << 3;
    }

    auto issue = [&](int k0, int bi) {
#pragma unroll
        for (int i = 0; i < 2; ++i)
            async_b128(&A[(size_t)(m0 + arow[i]) * lda + k0 + akc[i]],
                       &Als[bi][arow[i] * LDAB + akc[i]]);
#pragma unroll
        for (int i = 0; i < WNT; ++i)
            async_b128(&Wt[(size_t)(n0 + brow[i]) * ldb + k0 + bkc[i]],
                       &Bls[bi][brow[i] * LDAB + bkc[i]]);
    };

    v8f acc[2][2 * WNT];
#pragma unroll
    for (int i = 0; i < 2; ++i)
#pragma unroll
        for (int j = 0; j < 2 * WNT; ++j)
#pragma unroll
            for (int r = 0; r < 8; ++r) acc[i][j][r] = 0.f;

    const int nkt = K >> 5;
    issue(0, 0);
    for (int kt = 0; kt < nkt; ++kt) {
        wait_async0();          // own async copies (incl. tile kt) done
        __syncthreads();        // everyone's copies landed; prev-tile reads done
        if (kt + 1 < nkt) issue((kt + 1) << 5, (kt + 1) & 1);

        const __bf16* Ab = Als[kt & 1];
        const __bf16* Bb = Bls[kt & 1];
        v16bf af[2], bfr[2 * WNT];
#pragma unroll
        for (int ti = 0; ti < 2; ++ti)
            af[ti] = frag16(&Ab[(wm * 32 + ti * 16 + l16) * LDAB + kb8]);
#pragma unroll
        for (int tj = 0; tj < 2 * WNT; ++tj)
            bfr[tj] = frag16(&Bb[(wn * 32 * WNT + tj * 16 + l16) * LDAB + kb8]);

#pragma unroll
        for (int ti = 0; ti < 2; ++ti)
#pragma unroll
            for (int tj = 0; tj < 2 * WNT; ++tj)
                acc[ti][tj] = __builtin_amdgcn_wmma_f32_16x16x32_bf16(
                    false, af[ti], false, bfr[tj], (short)0, acc[ti][tj], false, false);
    }

#pragma unroll
    for (int ti = 0; ti < 2; ++ti)
#pragma unroll
        for (int tj = 0; tj < 2 * WNT; ++tj)
#pragma unroll
            for (int r = 0; r < 8; ++r) {
                int m = m0 + wm * 32 + ti * 16 + half * 8 + r;
                int n = n0 + wn * 32 * WNT + tj * 16 + l16;
                size_t idx = (size_t)m * N + n;
                float v = acc[ti][tj][r];
                if (mode == 2)       ((__bf16*)Cv)[idx] = (__bf16)v;
                else if (mode == 1)  ((float*)Cv)[idx] += v;
                else                 ((float*)Cv)[idx]  = v;
            }
}

// =====================================================================
// Flash attention: 1 wave per (b, h, 16-row q tile). Causal.
// Q/K/V bf16 in, O bf16 out.
// =====================================================================
__global__ __launch_bounds__(32)
void attn_flash_wmma(const __bf16* __restrict__ Q, const __bf16* __restrict__ Kp,
                     const __bf16* __restrict__ V, __bf16* __restrict__ O)
{
    __shared__ __align__(16) __bf16 Vls[DHd * 32];   // [dh][key]
    __shared__ __align__(16) __bf16 Pls[16 * 32];    // [row][key]

    const int lane = threadIdx.x;
    const int half = lane >> 4;
    const int l16  = lane & 15;
    const int kb8  = half * 8;
    const int q0   = blockIdx.x * 16;
    const int h    = blockIdx.y;
    const int b    = blockIdx.z;

    v16bf qf[4];
    const size_t qbase = ((size_t)b * Sd + q0 + l16) * Hd + h * DHd;
#pragma unroll
    for (int ks = 0; ks < 4; ++ks)
        qf[ks] = frag16(&Q[qbase + ks * 32 + kb8]);

    v8f o[8];
#pragma unroll
    for (int n = 0; n < 8; ++n)
#pragma unroll
        for (int r = 0; r < 8; ++r) o[n][r] = 0.f;
    float mrow[8], lrow[8];
#pragma unroll
    for (int r = 0; r < 8; ++r) { mrow[r] = -1e30f; lrow[r] = 0.f; }

    const int nkt = (q0 + 47) >> 5;
    for (int kt = 0; kt < nkt; ++kt) {
        const int kbase = kt * 32;

        // ---- stage V tile transposed [dh][key], packed-pair b32 stores ----
        const size_t vbase = ((size_t)b * Sd + kbase) * Hd + h * DHd;
#pragma unroll
        for (int i = 0; i < 8; ++i) {
            int pidx = lane + i * 32;          // 0..255
            int key2 = (pidx & 15) << 1;       // even key 0..30
            int dh8  = (pidx >> 4) << 3;       // 0..120
            v8bf a = *(const v8bf*)&V[vbase + (size_t)key2 * Hd + dh8];
            v8bf c = *(const v8bf*)&V[vbase + (size_t)(key2 + 1) * Hd + dh8];
#pragma unroll
            for (int j = 0; j < 8; ++j) {
                unsigned u = (unsigned)bfbits(a[j]) | ((unsigned)bfbits(c[j]) << 16);
                *(unsigned*)&Vls[(dh8 + j) * 32 + key2] = u;
            }
        }
        __syncthreads();

        // ---- scores: preload all K fragments, then 8 WMMAs ----
        const size_t kr0 = ((size_t)b * Sd + kbase + l16) * Hd + h * DHd;
        const size_t kr1 = kr0 + (size_t)16 * Hd;
        v16bf kf[8];
#pragma unroll
        for (int ks = 0; ks < 4; ++ks) {
            kf[ks]     = frag16(&Kp[kr0 + ks * 32 + kb8]);
            kf[4 + ks] = frag16(&Kp[kr1 + ks * 32 + kb8]);
        }
        v8f s0, s1;
#pragma unroll
        for (int r = 0; r < 8; ++r) { s0[r] = 0.f; s1[r] = 0.f; }
#pragma unroll
        for (int ks = 0; ks < 4; ++ks) {
            s0 = __builtin_amdgcn_wmma_f32_16x16x32_bf16(false, qf[ks], false, kf[ks],
                                                         (short)0, s0, false, false);
            s1 = __builtin_amdgcn_wmma_f32_16x16x32_bf16(false, qf[ks], false, kf[4 + ks],
                                                         (short)0, s1, false, false);
        }

        // ---- online softmax ----
#pragma unroll
        for (int r = 0; r < 8; ++r) {
            int row  = r + 8 * half;
            int qrow = q0 + row;
            int c0 = kbase + l16, c1 = kbase + 16 + l16;
            float v0 = (c0 <= qrow) ? s0[r] * QSCALE : -1e30f;
            float v1 = (c1 <= qrow) ? s1[r] * QSCALE : -1e30f;
            float mx = fmaxf(v0, v1);
#pragma unroll
            for (int off = 8; off >= 1; off >>= 1)
                mx = fmaxf(mx, __shfl_xor(mx, off, 32));
            float mnew = fmaxf(mrow[r], mx);
            float p0 = __expf(v0 - mnew);
            float p1 = __expf(v1 - mnew);
            float ps = p0 + p1;
#pragma unroll
            for (int off = 8; off >= 1; off >>= 1)
                ps += __shfl_xor(ps, off, 32);
            float corr = __expf(mrow[r] - mnew);
            lrow[r] = lrow[r] * corr + ps;
            mrow[r] = mnew;
#pragma unroll
            for (int n = 0; n < 8; ++n) o[n][r] *= corr;
            Pls[row * 32 + l16]      = (__bf16)p0;
            Pls[row * 32 + 16 + l16] = (__bf16)p1;
        }
        __syncthreads();

        // ---- O += P @ V ----
        v16bf pf = frag16(&Pls[l16 * 32 + kb8]);
#pragma unroll
        for (int n = 0; n < 8; ++n) {
            v16bf vf = frag16(&Vls[(n * 16 + l16) * 32 + kb8]);
            o[n] = __builtin_amdgcn_wmma_f32_16x16x32_bf16(false, pf, false, vf,
                                                           (short)0, o[n], false, false);
        }
        __syncthreads();
    }

#pragma unroll
    for (int r = 0; r < 8; ++r) {
        float inv = 1.f / lrow[r];
        size_t obase = ((size_t)b * Sd + q0 + r + 8 * half) * Hd + h * DHd;
#pragma unroll
        for (int n = 0; n < 8; ++n)
            O[obase + n * 16 + l16] = (__bf16)(o[n][r] * inv);
    }
}

// =====================================================================
// split_norm: per-row dual LayerNorm, bf16 output
// =====================================================================
__device__ __forceinline__ float blk_sum(float v, float* red) {
    int t = threadIdx.x;
    red[t] = v; __syncthreads();
    for (int o = 128; o > 0; o >>= 1) {
        if (t < o) red[t] += red[t + o];
        __syncthreads();
    }
    float r = red[0]; __syncthreads();
    return r;
}

__global__ __launch_bounds__(256)
void split_norm_kernel(const float* __restrict__ X, __bf16* __restrict__ Y,
                       const float* __restrict__ sg, const float* __restrict__ sb,
                       const float* __restrict__ rg, const float* __restrict__ rb)
{
    __shared__ float red[256];
    const int row = blockIdx.x;
    const int t = threadIdx.x;
    const float* x = X + (size_t)row * Hd;
    __bf16* y = Y + (size_t)row * Hd;

    float s = 0.f, q = 0.f;
    for (int c = t; c < SELd; c += 256) { float v = x[c]; s += v; q += v * v; }
    float sm = blk_sum(s, red), sq = blk_sum(q, red);
    float mu = sm / SELd;
    float rs = rsqrtf(sq / SELd - mu * mu + EPSf);
    for (int c = t; c < SELd; c += 256)
        y[c] = (__bf16)((x[c] - mu) * rs * sg[c] + sb[c]);

    s = 0.f; q = 0.f;
    for (int c = SELd + t; c < Hd; c += 256) { float v = x[c]; s += v; q += v * v; }
    sm = blk_sum(s, red); sq = blk_sum(q, red);
    const float NR = (float)(Hd - SELd);
    mu = sm / NR;
    rs = rsqrtf(sq / NR - mu * mu + EPSf);
    for (int c = SELd + t; c < Hd; c += 256)
        y[c] = (__bf16)((x[c] - mu) * rs * rg[c - SELd] + rb[c - SELd]);
}

// =====================================================================
// ssm_connection elementwise combine (fp32)
// =====================================================================
__global__ __launch_bounds__(256)
void ssm_combine_kernel(const float* __restrict__ X, const float* __restrict__ YP,
                        const float* __restrict__ DRAW, const float* __restrict__ Av,
                        const float* __restrict__ dub, float* __restrict__ OUT)
{
    size_t idx = (size_t)blockIdx.x * 256 + threadIdx.x;
    int c   = (int)(idx & (Hd - 1));
    int row = (int)(idx >> 11);
    float xv = X[idx], yv = YP[idx];
    if (c < SELd) {
        float an = -softplus_f(Av[c]);
        float d  = softplus_f(DRAW[(size_t)row * SELd + c] + dub[c]);
        float ab = __expf(d * an);
        float bb = (ab - 1.f) / an;
        OUT[idx] = ab * xv + bb * yv;
    } else {
        OUT[idx] = xv + yv;
    }
}

// =====================================================================
// silu(g)*u : bf16 in, bf16 out (in place over g)
// =====================================================================
__global__ __launch_bounds__(256)
void silu_mul_kernel(__bf16* __restrict__ G, const __bf16* __restrict__ U)
{
    size_t idx = (size_t)blockIdx.x * 256 + threadIdx.x;
    float g = (float)G[idx];
    float u = (float)U[idx];
    G[idx] = (__bf16)((g / (1.f + __expf(-g))) * u);
}

// =====================================================================
// host-side orchestration
// =====================================================================
static inline void launch_gemm(const __bf16* A, int lda, const __bf16* Wt, int ldb,
                               void* C, int M, int N, int K, int mode, hipStream_t s) {
    if (N % 128 == 0)
        gemm_bf16_wmma<2><<<dim3(N / 128, M / 128), 256, 0, s>>>(A, lda, Wt, ldb, C, M, N, K, mode);
    else
        gemm_bf16_wmma<1><<<dim3(N / 64, M / 128), 256, 0, s>>>(A, lda, Wt, ldb, C, M, N, K, mode);
}

static inline void launch_tconv(const float* W, __bf16* Wt, int K, int N, hipStream_t s) {
    transpose_conv_bf16<<<dim3(K / 32, N / 32), 256, 0, s>>>(W, Wt, K, N);
}

extern "C" void kernel_launch(void* const* d_in, const int* in_sizes, int n_in,
                              void* d_out, int out_size, void* d_ws, size_t ws_size,
                              hipStream_t stream) {
    const float* x        = (const float*)d_in[0];
    const float* an_sg    = (const float*)d_in[1];
    const float* an_sb    = (const float*)d_in[2];
    const float* an_rg    = (const float*)d_in[3];
    const float* an_rb    = (const float*)d_in[4];
    const float* mn_sg    = (const float*)d_in[5];
    const float* mn_sb    = (const float*)d_in[6];
    const float* mn_rg    = (const float*)d_in[7];
    const float* mn_rb    = (const float*)d_in[8];
    const float* Wq       = (const float*)d_in[9];
    const float* Wk       = (const float*)d_in[10];
    const float* Wv       = (const float*)d_in[11];
    const float* Wg       = (const float*)d_in[12];
    const float* Wu       = (const float*)d_in[13];
    const float* ac_A     = (const float*)d_in[14];
    const float* ac_W     = (const float*)d_in[15];
    const float* ac_dn_sg = (const float*)d_in[16];
    const float* ac_dn_sb = (const float*)d_in[17];
    const float* ac_dn_rg = (const float*)d_in[18];
    const float* ac_dn_rb = (const float*)d_in[19];
    const float* ac_dd    = (const float*)d_in[20];
    const float* ac_duw   = (const float*)d_in[21];
    const float* ac_dub   = (const float*)d_in[22];
    const float* mc_A     = (const float*)d_in[23];
    const float* mc_W     = (const float*)d_in[24];
    const float* mc_dn_sg = (const float*)d_in[25];
    const float* mc_dn_sb = (const float*)d_in[26];
    const float* mc_dn_rg = (const float*)d_in[27];
    const float* mc_dn_rb = (const float*)d_in[28];
    const float* mc_dd    = (const float*)d_in[29];
    const float* mc_duw   = (const float*)d_in[30];
    const float* mc_dub   = (const float*)d_in[31];
    float* out = (float*)d_out;

    const size_t NHE = (size_t)MROWS * Hd;
    const size_t NME = (size_t)MROWS * MLPd;
    char* p = (char*)d_ws;
    auto alloc = [&](size_t bytes) -> void* {
        void* r = (void*)p;
        p += (bytes + 255) & ~(size_t)255;
        return r;
    };
    __bf16* bxn   = (__bf16*)alloc(NHE * 2);
    __bf16* bq    = (__bf16*)alloc(NHE * 2);
    __bf16* bk    = (__bf16*)alloc(NHE * 2);
    __bf16* bv    = (__bf16*)alloc(NHE * 2);
    __bf16* battn = (__bf16*)alloc(NHE * 2);
    __bf16* bg    = (__bf16*)alloc(NME * 2);
    __bf16* bu    = (__bf16*)alloc(NME * 2);
    __bf16* bdmid = (__bf16*)alloc((size_t)MROWS * Rd * 2);
    float* f_yp    = (float*)alloc(NHE * 4);
    float* f_h1    = (float*)alloc(NHE * 4);
    float* f_dmid  = (float*)alloc((size_t)MROWS * Rd * 4);
    float* f_delta = (float*)alloc((size_t)MROWS * SELd * 4);
    __bf16* tWq   = (__bf16*)alloc((size_t)Hd * Hd * 2);
    __bf16* tWk   = (__bf16*)alloc((size_t)Hd * Hd * 2);
    __bf16* tWv   = (__bf16*)alloc((size_t)Hd * Hd * 2);
    __bf16* tWg   = (__bf16*)alloc((size_t)Hd * MLPd * 2);
    __bf16* tWu   = (__bf16*)alloc((size_t)Hd * MLPd * 2);
    __bf16* tacW  = (__bf16*)alloc((size_t)Hd * Hd * 2);
    __bf16* tmcW  = (__bf16*)alloc((size_t)MLPd * Hd * 2);
    __bf16* tacdd = (__bf16*)alloc((size_t)(2 * Hd) * Rd * 2);
    __bf16* tmcdd = (__bf16*)alloc((size_t)(Hd + MLPd) * Rd * 2);
    __bf16* tacdu = (__bf16*)alloc((size_t)Rd * SELd * 2);
    __bf16* tmcdu = (__bf16*)alloc((size_t)Rd * SELd * 2);

    launch_tconv(Wq,     tWq,   Hd,        Hd,   stream);
    launch_tconv(Wk,     tWk,   Hd,        Hd,   stream);
    launch_tconv(Wv,     tWv,   Hd,        Hd,   stream);
    launch_tconv(Wg,     tWg,   Hd,        MLPd, stream);
    launch_tconv(Wu,     tWu,   Hd,        MLPd, stream);
    launch_tconv(ac_W,   tacW,  Hd,        Hd,   stream);
    launch_tconv(mc_W,   tmcW,  MLPd,      Hd,   stream);
    launch_tconv(ac_dd,  tacdd, 2 * Hd,    Rd,   stream);
    launch_tconv(mc_dd,  tmcdd, Hd + MLPd, Rd,   stream);
    launch_tconv(ac_duw, tacdu, Rd,        SELd, stream);
    launch_tconv(mc_duw, tmcdu, Rd,        SELd, stream);

    const int snGrid   = MROWS;
    const int ssmGrid  = (int)(NHE / 256);
    const int siluGrid = (int)(NME / 256);
    const int dmGrid   = (int)((size_t)MROWS * Rd / 256);

    // ===== attention branch =====
    split_norm_kernel<<<snGrid, 256, 0, stream>>>(x, bxn, an_sg, an_sb, an_rg, an_rb);
    launch_gemm(bxn, Hd, tWq, Hd, bq, MROWS, Hd, Hd, 2, stream);
    launch_gemm(bxn, Hd, tWk, Hd, bk, MROWS, Hd, Hd, 2, stream);
    launch_gemm(bxn, Hd, tWv, Hd, bv, MROWS, Hd, Hd, 2, stream);
    attn_flash_wmma<<<dim3(Sd / 16, NHd, Bd), 32, 0, stream>>>(bq, bk, bv, battn);

    // ===== ssm_connection #1 =====
    launch_gemm(battn, Hd, tacW, Hd, f_yp, MROWS, Hd, Hd, 0, stream);
    split_norm_kernel<<<snGrid, 256, 0, stream>>>(x, bxn, ac_dn_sg, ac_dn_sb, ac_dn_rg, ac_dn_rb);
    launch_gemm(bxn,   Hd, tacdd,      2 * Hd, f_dmid, MROWS, Rd, Hd, 0, stream);
    launch_gemm(battn, Hd, tacdd + Hd, 2 * Hd, f_dmid, MROWS, Rd, Hd, 1, stream);
    conv_bf16<<<dmGrid, 256, 0, stream>>>(f_dmid, bdmid);
    launch_gemm(bdmid, Rd, tacdu, Rd, f_delta, MROWS, SELd, Rd, 0, stream);
    ssm_combine_kernel<<<ssmGrid, 256, 0, stream>>>(x, f_yp, f_delta, ac_A, ac_dub, f_h1);

    // ===== MLP branch =====
    split_norm_kernel<<<snGrid, 256, 0, stream>>>(f_h1, bxn, mn_sg, mn_sb, mn_rg, mn_rb);
    launch_gemm(bxn, Hd, tWg, Hd, bg, MROWS, MLPd, Hd, 2, stream);
    launch_gemm(bxn, Hd, tWu, Hd, bu, MROWS, MLPd, Hd, 2, stream);
    silu_mul_kernel<<<siluGrid, 256, 0, stream>>>(bg, bu);

    // ===== ssm_connection #2 =====
    launch_gemm(bg, MLPd, tmcW, MLPd, f_yp, MROWS, Hd, MLPd, 0, stream);
    split_norm_kernel<<<snGrid, 256, 0, stream>>>(f_h1, bxn, mc_dn_sg, mc_dn_sb, mc_dn_rg, mc_dn_rb);
    launch_gemm(bxn, Hd, tmcdd,       Hd + MLPd, f_dmid, MROWS, Rd, Hd,   0, stream);
    launch_gemm(bg, MLPd, tmcdd + Hd, Hd + MLPd, f_dmid, MROWS, Rd, MLPd, 1, stream);
    conv_bf16<<<dmGrid, 256, 0, stream>>>(f_dmid, bdmid);
    launch_gemm(bdmid, Rd, tmcdu, Rd, f_delta, MROWS, SELd, Rd, 0, stream);
    ssm_combine_kernel<<<ssmGrid, 256, 0, stream>>>(f_h1, f_yp, f_delta, mc_A, mc_dub, out);

    (void)in_sizes; (void)n_in; (void)out_size; (void)ws_size;
}
